// LieMultiHeadAttention_37881611551185
// MI455X (gfx1250) — compile-verified, matrix operand order checked by reading, not run on hardware
//
#include <hip/hip_runtime.h>
#include <hip/hip_bf16.h>

typedef __attribute__((ext_vector_type(16))) __bf16 v16bf;
typedef __attribute__((ext_vector_type(8)))  float  v8f;

#define DIM   768
#define NHEAD 12
#define HDIM  64
#define SEQ   1024
#define BATCH 4
#define MROWS (BATCH * SEQ)   // 4096
#define QKV3  (3 * DIM)       // 2304

// ---------------------------------------------------------------------------
// Fragment loader for 16-bit WMMA operands (CDNA5 16x16x32 layout).
// A-matrix 16x32: lane = M (0..15, halves), VGPR r holds packed K pair:
//   K = (r/4)*16 + (lane/16)*8 + (r%4)*2
// For the B operand of D = A*B with B = (row-major src)^T, the SAME loader
// applies with lane playing the N role (B[k][n] = src[n*ld + k]).
// Requires 4-byte-aligned base and even ld (all our strides are even).
// ---------------------------------------------------------------------------
__device__ __forceinline__ v16bf load_frag16(const __bf16* p, int ld) {
  union { v16bf v; unsigned u[8]; } f;
  const int lane = threadIdx.x & 31;
  const int m    = lane & 15;
  const int hf   = lane >> 4;
#pragma unroll
  for (int r = 0; r < 8; ++r) {
    const int k = ((r >> 2) << 4) + (hf << 3) + ((r & 3) << 1);
    f.u[r] = *(const unsigned*)(p + (size_t)m * ld + k);
  }
  return f.v;
}

__device__ __forceinline__ v8f wmma_bf16(v16bf a, v16bf b, v8f c) {
  return __builtin_amdgcn_wmma_f32_16x16x32_bf16(
      /*neg_a=*/false, a, /*neg_b=*/false, b,
      /*c_mod=*/(short)0, c, /*reuse_a=*/false, /*reuse_b=*/false);
}

// ---------------------------------------------------------------------------
// Converters
// ---------------------------------------------------------------------------
__global__ void k_cvt(const float* __restrict__ src, __bf16* __restrict__ dst, int n) {
  int i = blockIdx.x * 256 + threadIdx.x;
  if (i < n) dst[i] = (__bf16)src[i];
}

// w: (K x N) row-major fp32  ->  wt: (N x K) row-major bf16
__global__ void k_cvt_t(const float* __restrict__ w, __bf16* __restrict__ wt, int K, int N) {
  int i = blockIdx.x * 256 + threadIdx.x;
  if (i < K * N) {
    int k = i / N, n = i % N;
    wt[(size_t)n * K + k] = (__bf16)w[i];
  }
}

// ---------------------------------------------------------------------------
// QKV projection: (4096 x 768) @ (768 x 2304) + b, scatter into per-head
// bf16 buffers: q (B,H,N,HD) pre-scaled by HD^-0.5, k (B,H,N,HD),
// v transposed (B,H,HD,N) so it serves directly as a WMMA B operand.
// grid = (256, 18), block = 256 (8 waves, one 16x16 tile each)
// ---------------------------------------------------------------------------
__global__ __launch_bounds__(256) void k_qkv(
    const __bf16* __restrict__ xb, const __bf16* __restrict__ wt,
    const float* __restrict__ qkv_b,
    __bf16* __restrict__ qb, __bf16* __restrict__ kb, __bf16* __restrict__ vtb) {
  const int K = DIM;
  const int wave = threadIdx.x >> 5;
  const int m0 = blockIdx.x * 16;
  const int c0 = (blockIdx.y * 8 + wave) * 16;

  const __bf16* ap = xb + (size_t)m0 * K;
  const __bf16* bp = wt + (size_t)c0 * K;
  v8f acc = {};
#pragma unroll 4
  for (int k0 = 0; k0 < K; k0 += 32) {
    v16bf a = load_frag16(ap + k0, K);
    v16bf b = load_frag16(bp + k0, K);
    acc = wmma_bf16(a, b, acc);
  }

  const int lane = threadIdx.x & 31;
  const int coln = lane & 15;
  const int hf   = lane >> 4;
#pragma unroll
  for (int r = 0; r < 8; ++r) {
    int row = m0 + r + 8 * hf;            // [0,4096)
    int c   = c0 + coln;                  // [0,2304)
    float v = acc[r] + qkv_b[c];
    int bidx = row >> 10, n = row & (SEQ - 1);
    int which = c / DIM, cc = c % DIM;
    int h = cc >> 6, d = cc & (HDIM - 1);
    size_t bh = (size_t)(bidx * NHEAD + h);
    if (which == 0)      qb[(bh * SEQ + n) * HDIM + d] = (__bf16)(v * 0.125f); // *HD^-0.5
    else if (which == 1) kb[(bh * SEQ + n) * HDIM + d] = (__bf16)v;
    else                 vtb[(bh * HDIM + d) * SEQ + n] = (__bf16)v;
  }
}

// ---------------------------------------------------------------------------
// Fused attention block: logits (WMMA) + lie-bias MLP + softmax + attn store
// + O = P @ V (WMMA).   grid = (64 row-tiles, 48 b*h), block = 256.
// Dynamic LDS: S fp32 16x1024 (64KB) | P bf16 16x1024 (32KB) | Oacc 16x64 (4KB)
// ---------------------------------------------------------------------------
__global__ __launch_bounds__(256) void k_attn(
    const __bf16* __restrict__ qb, const __bf16* __restrict__ kb,
    const __bf16* __restrict__ vtb, const float* __restrict__ bias,
    const float* __restrict__ w1, const float* __restrict__ b1,
    const float* __restrict__ w2, const float* __restrict__ b2,
    float* __restrict__ attn_out, __bf16* __restrict__ obuf) {
  extern __shared__ char smem[];
  float*  S    = (float*)smem;                       // 16*1024 fp32
  __bf16* P    = (__bf16*)(smem + 16 * SEQ * 4);     // 16*1024 bf16
  float*  Oacc = (float*)(smem + 16 * SEQ * 4 + 16 * SEQ * 2); // 16*64 fp32

  const int bh = blockIdx.y;                  // b*12 + h
  const int b_ = bh / NHEAD, h = bh % NHEAD;
  const int i0 = blockIdx.x * 16;             // query row tile
  const int wave = threadIdx.x >> 5;
  const int lane = threadIdx.x & 31;
  const int coln = lane & 15;
  const int hf   = lane >> 4;

  const __bf16* qp = qb  + ((size_t)bh * SEQ + i0) * HDIM;
  const __bf16* kp = kb  + (size_t)bh * SEQ * HDIM;
  const __bf16* vp = vtb + (size_t)bh * HDIM * SEQ;

  // MLP params for this head (broadcast loads -> L2/constant path)
  float w1r[16], b1r[16], w2r[16];
#pragma unroll
  for (int t = 0; t < 16; ++t) {
    w1r[t] = w1[t]; b1r[t] = b1[t]; w2r[t] = w2[t * NHEAD + h];
  }
  const float b2h = b2[h];

  // Q fragments (HD=64 -> two k-steps of 32)
  v16bf qa0 = load_frag16(qp, HDIM);
  v16bf qa1 = load_frag16(qp + 32, HDIM);

  // Logit strip: each wave owns 8 column tiles (j-range of 128)
  for (int t = 0; t < 8; ++t) {
    const int j0 = (wave * 8 + t) * 16;
    v16bf kf0 = load_frag16(kp + (size_t)j0 * HDIM, HDIM);
    v16bf kf1 = load_frag16(kp + (size_t)j0 * HDIM + 32, HDIM);
    v8f acc = {};
    acc = wmma_bf16(qa0, kf0, acc);
    acc = wmma_bf16(qa1, kf1, acc);
#pragma unroll
    for (int r = 0; r < 8; ++r) {
      const int row = r + 8 * hf;              // local query row 0..15
      const int j   = j0 + coln;
      const float bv = bias[((size_t)b_ * SEQ + (i0 + row)) * SEQ + j];
      float lie = b2h;
#pragma unroll
      for (int tt = 0; tt < 16; ++tt) {
        float hd = fmaf(bv, w1r[tt], b1r[tt]);
        hd = fmaxf(hd, 0.0f);
        lie = fmaf(hd, w2r[tt], lie);
      }
      S[row * SEQ + j] = acc[r] + lie;
    }
  }
  __syncthreads();

  // Softmax: wave w handles rows 2w, 2w+1 (wave32 shuffle reductions)
  for (int rr = 0; rr < 2; ++rr) {
    const int row = wave * 2 + rr;
    float mx = -3.4e38f;
    for (int j = lane; j < SEQ; j += 32) mx = fmaxf(mx, S[row * SEQ + j]);
#pragma unroll
    for (int o = 16; o > 0; o >>= 1) mx = fmaxf(mx, __shfl_xor(mx, o, 32));
    float sum = 0.0f;
    for (int j = lane; j < SEQ; j += 32) {
      float e = __expf(S[row * SEQ + j] - mx);
      S[row * SEQ + j] = e;
      sum += e;
    }
#pragma unroll
    for (int o = 16; o > 0; o >>= 1) sum += __shfl_xor(sum, o, 32);
    const float inv = 1.0f / sum;
    float* arow = attn_out + ((size_t)bh * SEQ + (i0 + row)) * SEQ;
    for (int j = lane; j < SEQ; j += 32) {
      float p = S[row * SEQ + j] * inv;
      arow[j] = p;                       // fp32 attn output (required)
      P[row * SEQ + j] = (__bf16)p;      // bf16 operand for P@V
    }
  }
  for (int idx = threadIdx.x; idx < 16 * HDIM; idx += 256) Oacc[idx] = 0.0f;
  __syncthreads();

  // O(16x64) = P(16x1024) @ V(1024x64): wave w covers k in [128w,128w+128),
  // partials reduced with LDS float atomics (ds_add_f32).
  for (int dt = 0; dt < 4; ++dt) {
    const int d0 = dt * 16;
    v8f acc = {};
#pragma unroll
    for (int kk = 0; kk < 4; ++kk) {
      const int k0 = wave * 128 + kk * 32;
      v16bf a = load_frag16(P + k0, SEQ);                    // P[m][k0+k]
      v16bf b = load_frag16(vp + (size_t)d0 * SEQ + k0, SEQ); // V[k][d0+n]
      acc = wmma_bf16(a, b, acc);
    }
#pragma unroll
    for (int r = 0; r < 8; ++r) {
      const int row = r + 8 * hf;
      atomicAdd(&Oacc[row * HDIM + d0 + coln], acc[r]);
    }
  }
  __syncthreads();

  // Store O tile as bf16 in (B,N,C) layout for the projection GEMM
  for (int idx = threadIdx.x; idx < 16 * HDIM; idx += 256) {
    const int row = idx >> 6, d = idx & (HDIM - 1);
    obuf[((size_t)(b_ * SEQ + i0 + row)) * DIM + h * HDIM + d] = (__bf16)Oacc[idx];
  }
}

// ---------------------------------------------------------------------------
// Output projection: (4096 x 768) @ (768 x 768) + b -> fp32 d_out
// grid = (256, 6), block = 256
// ---------------------------------------------------------------------------
__global__ __launch_bounds__(256) void k_proj(
    const __bf16* __restrict__ ob, const __bf16* __restrict__ wt,
    const float* __restrict__ pb, float* __restrict__ out) {
  const int K = DIM;
  const int wave = threadIdx.x >> 5;
  const int m0 = blockIdx.x * 16;
  const int c0 = (blockIdx.y * 8 + wave) * 16;
  v8f acc = {};
#pragma unroll 4
  for (int k0 = 0; k0 < K; k0 += 32) {
    v16bf a = load_frag16(ob + (size_t)m0 * K + k0, K);
    v16bf b = load_frag16(wt + (size_t)c0 * K + k0, K);
    acc = wmma_bf16(a, b, acc);
  }
  const int lane = threadIdx.x & 31;
  const int coln = lane & 15;
  const int hf   = lane >> 4;
#pragma unroll
  for (int r = 0; r < 8; ++r) {
    const int row = m0 + r + 8 * hf;
    const int c   = c0 + coln;
    out[(size_t)row * DIM + c] = acc[r] + pb[c];
  }
}

// ---------------------------------------------------------------------------
extern "C" void kernel_launch(void* const* d_in, const int* in_sizes, int n_in,
                              void* d_out, int out_size, void* d_ws, size_t ws_size,
                              hipStream_t stream) {
  const float* x      = (const float*)d_in[0];
  const float* bias   = (const float*)d_in[1];
  const float* qkv_w  = (const float*)d_in[2];
  const float* qkv_b  = (const float*)d_in[3];
  const float* mlp_w1 = (const float*)d_in[4];
  const float* mlp_b1 = (const float*)d_in[5];
  const float* mlp_w2 = (const float*)d_in[6];
  const float* mlp_b2 = (const float*)d_in[7];
  const float* proj_w = (const float*)d_in[8];
  const float* proj_b = (const float*)d_in[9];

  float* out  = (float*)d_out;                              // (B,N,C)
  float* attn = out + (size_t)MROWS * DIM;                  // (B,H,N,N)

  // Workspace partition (bf16 buffers), ~36 MB total
  char* ws = (char*)d_ws;
  size_t off = 0;
  auto take = [&](size_t bytes) { char* p = ws + off; off += (bytes + 255) & ~(size_t)255; return p; };
  __bf16* xb     = (__bf16*)take((size_t)MROWS * DIM * 2);
  __bf16* wqkvT  = (__bf16*)take((size_t)QKV3 * DIM * 2);
  __bf16* wprojT = (__bf16*)take((size_t)DIM * DIM * 2);
  __bf16* qb     = (__bf16*)take((size_t)BATCH * NHEAD * SEQ * HDIM * 2);
  __bf16* kb     = (__bf16*)take((size_t)BATCH * NHEAD * SEQ * HDIM * 2);
  __bf16* vtb    = (__bf16*)take((size_t)BATCH * NHEAD * HDIM * SEQ * 2);
  __bf16* obuf   = (__bf16*)take((size_t)MROWS * DIM * 2);
  (void)in_sizes; (void)n_in; (void)out_size; (void)ws_size;

  const int nx = MROWS * DIM;
  k_cvt<<<(nx + 255) / 256, 256, 0, stream>>>(x, xb, nx);
  k_cvt_t<<<(DIM * QKV3 + 255) / 256, 256, 0, stream>>>(qkv_w, wqkvT, DIM, QKV3);
  k_cvt_t<<<(DIM * DIM + 255) / 256, 256, 0, stream>>>(proj_w, wprojT, DIM, DIM);

  k_qkv<<<dim3(MROWS / 16, QKV3 / 16 / 8), 256, 0, stream>>>(xb, wqkvT, qkv_b, qb, kb, vtb);

  const size_t smem = (size_t)16 * SEQ * 4 + (size_t)16 * SEQ * 2 + (size_t)16 * HDIM * 4; // 100 KB < 320 KB/WGP
  k_attn<<<dim3(SEQ / 16, BATCH * NHEAD), 256, smem, stream>>>(
      qb, kb, vtb, bias, mlp_w1, mlp_b1, mlp_w2, mlp_b2, attn, obuf);

  k_proj<<<dim3(MROWS / 16, DIM / 16 / 8), 256, 0, stream>>>(obuf, wprojT, proj_b, out);
}
